// TransformerBlock_87892210745578
// MI455X (gfx1250) — matrix operand order repaired and checked by reading, not compiled, not run on hardware
//
#include <hip/hip_runtime.h>

// ---------------------------------------------------------------------------
// Qwen3-style transformer block for MI455X (gfx1250, wave32, WMMA bf16).
// - All GEMMs: v_wmma_f32_16x16x32_bf16, f32 accumulate.
// - Weights pre-converted to TRANSPOSED bf16 mirrors (B^T[N][K]) so every
//   B fragment is two 128-bit loads (K contiguous per lane).
// - B panels staged to LDS via gfx1250 async global->LDS (ASYNCcnt),
//   double buffered, shared by 8 waves per block.
// - Flash attention with online softmax; V stored transposed per (b,h).
// ---------------------------------------------------------------------------

#define BB 2
#define SS 1024
#define DM 2048
#define HH 16
#define HD 128
#define EE 8
#define MHID 768
#define TT (BB * SS)

typedef __attribute__((ext_vector_type(16))) __bf16 v16bf;
typedef __attribute__((ext_vector_type(8)))  float  v8f;

#define AS1 __attribute__((address_space(1)))
#define AS3 __attribute__((address_space(3)))
typedef int v4i_b __attribute__((vector_size(16)));   // matches builtin param type

#if defined(__has_builtin)
#  if __has_builtin(__builtin_amdgcn_global_load_async_to_lds_b128) && \
      __has_builtin(__builtin_amdgcn_s_wait_asynccnt)
#    define USE_ASYNC_LDS 1
#  endif
#endif
#ifndef USE_ASYNC_LDS
#  define USE_ASYNC_LDS 0
#endif

// 16B global -> LDS stage (async on CDNA5, falls back to reg round-trip).
__device__ inline void stage_b128(const __bf16* g, __bf16* l) {
#if USE_ASYNC_LDS
  __builtin_amdgcn_global_load_async_to_lds_b128(
      (AS1 v4i_b*)g, (AS3 v4i_b*)l, /*offset=*/0, /*cpol=*/0);
#else
  *reinterpret_cast<uint4*>(l) = *reinterpret_cast<const uint4*>(g);
#endif
}
__device__ inline void stage_wait() {
#if USE_ASYNC_LDS
  __builtin_amdgcn_s_wait_asynccnt(0);   // s_wait_asynccnt 0
#endif
  __syncthreads();
}

// ---------------------------------------------------------------------------
// WMMA wrapper
// ---------------------------------------------------------------------------
__device__ inline v8f wmma_bf16(v16bf a, v16bf b, v8f c) {
  return __builtin_amdgcn_wmma_f32_16x16x32_bf16(
      /*neg_a=*/false, a, /*neg_b=*/false, b,
      /*c_mod=*/(short)0, c, /*reuse_a=*/false, /*reuse_b=*/false);
}

// ---------------------------------------------------------------------------
// Fragment loaders (layouts per CDNA5 ISA §7.12.2, wave32)
// A 16x32 bf16: lanes 0-15 row M=lane, K = 0..7 & 16..23 ; lanes 16-31 row
// M=lane-16, K = 8..15 & 24..31.  Two 16B loads per lane.
// ---------------------------------------------------------------------------
__device__ inline v16bf load_fragA_k32(const __bf16* p, int stride) {
  const int lane = threadIdx.x & 31;
  const int row  = lane & 15;
  const int lo   = (lane >> 4) << 3;          // 0 or 8
  const __bf16* q = p + (size_t)row * stride + lo;
  union { uint4 u; __bf16 h[8]; } c0, c1;
  c0.u = *reinterpret_cast<const uint4*>(q);
  c1.u = *reinterpret_cast<const uint4*>(q + 16);
  v16bf f;
#pragma unroll
  for (int i = 0; i < 8; ++i) { f[i] = c0.h[i]; f[i + 8] = c1.h[i]; }
  return f;
}

// B 32x16 bf16: lane = column N (lane&15); lanes 0-15 hold K=0..7 & 16..23,
// lanes 16-31 hold K=8..15 & 24..31.  Source layout: element (k,n) at p[n*stride + k]
// (K contiguous per column) -> two 16B loads per lane.
__device__ inline v16bf load_fragB_kcontig(const __bf16* p, int stride) {
  const int lane = threadIdx.x & 31;
  const int n  = lane & 15;
  const int lo = (lane >> 4) << 3;            // 0 or 8
  const __bf16* q = p + (size_t)n * stride + lo;
  union { uint4 u; __bf16 h[8]; } c0, c1;
  c0.u = *reinterpret_cast<const uint4*>(q);
  c1.u = *reinterpret_cast<const uint4*>(q + 16);
  v16bf f;
#pragma unroll
  for (int i = 0; i < 8; ++i) { f[i] = c0.h[i]; f[i + 8] = c1.h[i]; }
  return f;
}

// ---------------------------------------------------------------------------
// GEMM: C[M,N](f32) = A[M,K](bf16) * B[K,N](bf16), with B given TRANSPOSED:
// Bt[N][K] row-major. Block = 256 threads (8 waves); block tile 128x64.
// B panel = 64(N) x 32(K) bf16 (4 KB) in LDS, K contiguous per N row, double
// buffered, async-staged, shared by all 8 waves.
// Requires M%128==0, N%64==0, K%32==0 (true for all call sites here).
// ---------------------------------------------------------------------------
__global__ __launch_bounds__(256)
void gemm_bf16_kernel(const __bf16* __restrict__ A, const __bf16* __restrict__ Bt,
                      float* __restrict__ C, const float* __restrict__ resid,
                      const float* __restrict__ rowscale, int rs_stride,
                      int M, int N, int K, int accum) {
  __shared__ __align__(16) __bf16 bp[2][64 * 32];

  const int blocks_n = N >> 6;
  const int bm = blockIdx.x / blocks_n;
  const int bn = blockIdx.x % blocks_n;
  const int n0 = bn << 6;
  const int wave = threadIdx.x >> 5;
  const int m0 = (bm << 7) + (wave << 4);
  const int lane = threadIdx.x & 31;

  // Staging coords: 256 threads x 16B = one 64x32 bf16 panel.
  const int srow = threadIdx.x >> 2;          // N row 0..63
  const int schk = (threadIdx.x & 3) << 3;    // K chunk 0,8,16,24
  const __bf16* Bp = Bt + (size_t)(n0 + srow) * K + schk;  // + k0
  __bf16* lp = &bp[0][srow * 32 + schk];

  v8f acc[4];
#pragma unroll
  for (int f = 0; f < 4; ++f) { v8f z = {}; acc[f] = z; }

  stage_b128(Bp, lp);
  int cur = 0;
  for (int k0 = 0; k0 < K; k0 += 32) {
    stage_wait();  // panel 'cur' resident; all waves done reading the other one
    if (k0 + 32 < K)
      stage_b128(Bp + (k0 + 32), &bp[cur ^ 1][srow * 32 + schk]);
    v16bf a = load_fragA_k32(A + (size_t)m0 * K + k0, K);
#pragma unroll
    for (int f = 0; f < 4; ++f) {
      v16bf b = load_fragB_kcontig(&bp[cur][(f * 16) * 32], 32);
      acc[f] = wmma_bf16(a, b, acc[f]);
    }
    cur ^= 1;
  }

  // C layout: lane holds column (lane&15); rows m0 + r + 8*(lane>=16).
  const int col   = lane & 15;
  const int rbase = (lane >> 4) << 3;
#pragma unroll
  for (int f = 0; f < 4; ++f) {
#pragma unroll
    for (int r = 0; r < 8; ++r) {
      const int row = m0 + rbase + r;
      const size_t idx = (size_t)row * N + n0 + f * 16 + col;
      float v = acc[f][r];
      if (rowscale) v *= rowscale[(size_t)row * rs_stride];
      if (resid)    v += resid[idx];
      if (accum)    v += C[idx];
      C[idx] = v;
    }
  }
}

// ---------------------------------------------------------------------------
// Flash attention: one wave per 16-query tile per (b,h). Online softmax,
// P re-shaped through per-wave LDS into an A fragment, P*V via WMMA.
// V is pre-transposed: vT[b][hd (=h*HD+d)][s].
// ---------------------------------------------------------------------------
__global__ __launch_bounds__(128)
void flash_attn_kernel(const __bf16* __restrict__ qb, const __bf16* __restrict__ kb,
                       const __bf16* __restrict__ vT, __bf16* __restrict__ ob) {
  const int nqt = SS / 16;
  const int gw  = (blockIdx.x * blockDim.x + threadIdx.x) >> 5;
  if (gw >= BB * HH * nqt) return;
  const int qt = gw % nqt;
  const int bh = gw / nqt;
  const int h  = bh % HH;
  const int b  = bh / HH;
  const int q0 = qt * 16;

  const int lane  = threadIdx.x & 31;
  const int wid   = threadIdx.x >> 5;
  const int col   = lane & 15;
  const int rhalf = (lane >> 4) << 3;

  __shared__ __align__(16) __bf16 pls[4][16 * 32];

  const size_t rowbase = ((size_t)b * SS) * DM + (size_t)h * HD;
  const size_t vbase   = ((size_t)b * DM + (size_t)h * HD) * SS;  // vT[b][h*HD][0]

  v16bf qfr[4];
#pragma unroll
  for (int f = 0; f < 4; ++f)
    qfr[f] = load_fragA_k32(qb + rowbase + (size_t)q0 * DM + f * 32, DM);

  v8f o[8];
#pragma unroll
  for (int f = 0; f < 8; ++f) { v8f z = {}; o[f] = z; }
  float mrow[8], lrow[8];
#pragma unroll
  for (int r = 0; r < 8; ++r) { mrow[r] = -1e30f; lrow[r] = 0.f; }

  const float scale = 0.08838834764831845f;     // 1/sqrt(128)
  const int jend = q0 + 15;

  for (int j0 = 0; j0 <= jend; j0 += 32) {
    if (j0 + 32 <= jend)  // warm cache for the next K tile
      __builtin_prefetch(kb + rowbase + (size_t)(j0 + 32) * DM, 0, 1);
    v8f s[2];
#pragma unroll
    for (int sub = 0; sub < 2; ++sub) {
      const int kbase = j0 + sub * 16;
      v8f a = {};
      if (kbase <= jend) {
#pragma unroll
        for (int f = 0; f < 4; ++f) {
          v16bf bf = load_fragB_kcontig(kb + rowbase + (size_t)kbase * DM + f * 32, DM);
          a = wmma_bf16(qfr[f], bf, a);
        }
      }
#pragma unroll
      for (int r = 0; r < 8; ++r) {
        const int qrow = q0 + rhalf + r;
        const int key  = kbase + col;
        float v = a[r] * scale;
        if (key > qrow || kbase > jend) v = -1e30f;
        a[r] = v;
      }
      s[sub] = a;
    }

    float mt[8];
#pragma unroll
    for (int r = 0; r < 8; ++r) {
      float m = fmaxf(s[0][r], s[1][r]);
#pragma unroll
      for (int off = 8; off >= 1; off >>= 1) m = fmaxf(m, __shfl_xor(m, off));
      mt[r] = m;
    }

    float scl[8];
#pragma unroll
    for (int r = 0; r < 8; ++r) {
      const float mn = fmaxf(mrow[r], mt[r]);
      scl[r]  = __expf(mrow[r] - mn);
      mrow[r] = mn;
    }

    float lt[8];
#pragma unroll
    for (int r = 0; r < 8; ++r) lt[r] = 0.f;
#pragma unroll
    for (int sub = 0; sub < 2; ++sub) {
#pragma unroll
      for (int r = 0; r < 8; ++r) {
        const float p = __expf(s[sub][r] - mrow[r]);
        pls[wid][(rhalf + r) * 32 + sub * 16 + col] = (__bf16)p;
        lt[r] += p;
      }
    }
#pragma unroll
    for (int r = 0; r < 8; ++r) {
      float t = lt[r];
#pragma unroll
      for (int off = 8; off >= 1; off >>= 1) t += __shfl_xor(t, off);
      lrow[r] = lrow[r] * scl[r] + t;
    }

#pragma unroll
    for (int f = 0; f < 8; ++f)
#pragma unroll
      for (int r = 0; r < 8; ++r) o[f][r] *= scl[r];

    // LDS ops are in-order per wave -> safe same-wave store->load.
    v16bf pa = load_fragA_k32(&pls[wid][0], 32);
#pragma unroll
    for (int f = 0; f < 8; ++f) {
      // V fragment: element (kk, n) = V[j0+kk, hd0+n] = vT[hd0+n][j0+kk]
      v16bf vf = load_fragB_kcontig(vT + vbase + (size_t)(f * 16) * SS + j0, SS);
      o[f] = wmma_bf16(pa, vf, o[f]);
    }
  }

#pragma unroll
  for (int f = 0; f < 8; ++f) {
#pragma unroll
    for (int r = 0; r < 8; ++r) {
      const float inv = 1.f / lrow[r];
      ob[rowbase + (size_t)(q0 + rhalf + r) * DM + f * 16 + col] =
          (__bf16)(o[f][r] * inv);
    }
  }
}

// ---------------------------------------------------------------------------
// RMSNorm over rows of length DM. Writes bf16 (always) and f32 (optional).
// ---------------------------------------------------------------------------
__global__ void rmsnorm_kernel(const float* __restrict__ X, const float* __restrict__ W,
                               __bf16* __restrict__ OB, float* __restrict__ OF) {
  const int t = blockIdx.x;
  const float* row = X + (size_t)t * DM;
  float ss = 0.f;
  for (int i = threadIdx.x; i < DM; i += blockDim.x) { float v = row[i]; ss += v * v; }
#pragma unroll
  for (int off = 16; off >= 1; off >>= 1) ss += __shfl_xor(ss, off);
  __shared__ float red[8];
  if ((threadIdx.x & 31) == 0) red[threadIdx.x >> 5] = ss;
  __syncthreads();
  float tot = 0.f;
#pragma unroll
  for (int i = 0; i < 8; ++i) tot += red[i];
  const float rs = rsqrtf(tot / (float)DM + 1e-6f);
  for (int i = threadIdx.x; i < DM; i += blockDim.x) {
    const float v = row[i] * rs * W[i];
    OB[(size_t)t * DM + i] = (__bf16)v;
    if (OF) OF[(size_t)t * DM + i] = v;
  }
}

// ---------------------------------------------------------------------------
// Per-head q/k RMSNorm + RoPE (rotate_half). One block (128 threads) per (t,h).
// ---------------------------------------------------------------------------
__global__ void qknorm_rope_kernel(const float* __restrict__ QF, const float* __restrict__ KF,
                                   const float* __restrict__ QW, const float* __restrict__ KW,
                                   const float* __restrict__ ROPE,
                                   __bf16* __restrict__ QB, __bf16* __restrict__ KB) {
  const int th = blockIdx.x;
  const int t  = th >> 4;
  const int h  = th & 15;
  const int s  = t & (SS - 1);
  const int i  = threadIdx.x;
  __shared__ float buf[HD];
  __shared__ float red[4];
  const float cs = ROPE[(size_t)s * (2 * HD) + i];
  const float sn = ROPE[(size_t)s * (2 * HD) + HD + i];
  const size_t base = (size_t)t * DM + (size_t)h * HD;
#pragma unroll
  for (int which = 0; which < 2; ++which) {
    const float* src = which ? KF : QF;
    const float* w   = which ? KW : QW;
    __bf16*      dst = which ? KB : QB;
    const float v = src[base + i];
    float ss2 = v * v;
#pragma unroll
    for (int off = 16; off >= 1; off >>= 1) ss2 += __shfl_xor(ss2, off);
    if ((i & 31) == 0) red[i >> 5] = ss2;
    __syncthreads();
    const float tot = red[0] + red[1] + red[2] + red[3];
    const float n = v * rsqrtf(tot / (float)HD + 1e-6f) * w[i];
    buf[i] = n;
    __syncthreads();
    const float rot = (i < 64) ? -buf[i + 64] : buf[i - 64];
    dst[base + i] = (__bf16)(n * cs + rot * sn);
    __syncthreads();
  }
}

// ---------------------------------------------------------------------------
// Tiled transpose + f32->bf16: out[z][c][r] = (bf16) in[z][r][c].
// Grid: (C/32, R/32, Z), block (32, 8). R, C multiples of 32.
// ---------------------------------------------------------------------------
__global__ void cvt_transpose_kernel(const float* __restrict__ in, __bf16* __restrict__ out,
                                     int R, int C) {
  __shared__ float tile[32][33];
  in  += (size_t)blockIdx.z * R * C;
  out += (size_t)blockIdx.z * R * C;
  const int c0 = blockIdx.x * 32;
  const int r0 = blockIdx.y * 32;
#pragma unroll
  for (int dy = threadIdx.y; dy < 32; dy += 8)
    tile[dy][threadIdx.x] = in[(size_t)(r0 + dy) * C + c0 + threadIdx.x];
  __syncthreads();
#pragma unroll
  for (int dy = threadIdx.y; dy < 32; dy += 8)
    out[(size_t)(c0 + dy) * R + r0 + threadIdx.x] = (__bf16)tile[threadIdx.x][dy];
}

// ---------------------------------------------------------------------------
// Elementwise helpers
// ---------------------------------------------------------------------------
__global__ void copy_f32_kernel(const float* __restrict__ in, float* __restrict__ out, int n) {
  const int i = blockIdx.x * blockDim.x + threadIdx.x;
  if (i < n) out[i] = in[i];
}

__global__ void silu_mul_kernel(const float* __restrict__ G, const float* __restrict__ U,
                                __bf16* __restrict__ O, int n) {
  const int i = blockIdx.x * blockDim.x + threadIdx.x;
  if (i < n) {
    const float g = G[i];
    O[i] = (__bf16)((g / (1.f + __expf(-g))) * U[i]);
  }
}

// ---------------------------------------------------------------------------
// MoE gating: logits = x2 @ gate_w (D x 8), softmax, top-2, renormalize.
// ---------------------------------------------------------------------------
__global__ void gate_kernel(const float* __restrict__ x2, const float* __restrict__ gw,
                            float* __restrict__ wts) {
  const int t = blockIdx.x;
  const float* row = x2 + (size_t)t * DM;
  float part[EE];
#pragma unroll
  for (int e = 0; e < EE; ++e) part[e] = 0.f;
  for (int d = threadIdx.x; d < DM; d += blockDim.x) {
    const float xv = row[d];
#pragma unroll
    for (int e = 0; e < EE; ++e) part[e] += xv * gw[(size_t)d * EE + e];
  }
#pragma unroll
  for (int e = 0; e < EE; ++e)
#pragma unroll
    for (int off = 16; off >= 1; off >>= 1) part[e] += __shfl_xor(part[e], off);
  __shared__ float red[8][EE];
  const int wid = threadIdx.x >> 5;
  if ((threadIdx.x & 31) == 0)
#pragma unroll
    for (int e = 0; e < EE; ++e) red[wid][e] = part[e];
  __syncthreads();
  if (threadIdx.x == 0) {
    float l[EE];
#pragma unroll
    for (int e = 0; e < EE; ++e) {
      float s = 0.f;
#pragma unroll
      for (int w = 0; w < 8; ++w) s += red[w][e];
      l[e] = s;
    }
    int i1 = 0;
#pragma unroll
    for (int e = 1; e < EE; ++e) if (l[e] > l[i1]) i1 = e;
    int i2 = (i1 == 0) ? 1 : 0;
#pragma unroll
    for (int e = 0; e < EE; ++e) if (e != i1 && l[e] > l[i2]) i2 = e;
    const float r = __expf(l[i2] - l[i1]);
    const float w1 = 1.f / (1.f + r);
    const float w2 = r / (1.f + r);
#pragma unroll
    for (int e = 0; e < EE; ++e)
      wts[(size_t)t * EE + e] = (e == i1) ? w1 : ((e == i2) ? w2 : 0.f);
  }
}

// ---------------------------------------------------------------------------
// Host-side launch
// ---------------------------------------------------------------------------
static void launch_gemm(const __bf16* A, const __bf16* Bt, float* C,
                        const float* resid, const float* rowscale, int rs_stride,
                        int M, int N, int K, int accum, hipStream_t stream) {
  const int blocks = (M / 128) * (N / 64);
  gemm_bf16_kernel<<<blocks, 256, 0, stream>>>(A, Bt, C, resid, rowscale, rs_stride,
                                               M, N, K, accum);
}

extern "C" void kernel_launch(void* const* d_in, const int* in_sizes, int n_in,
                              void* d_out, int out_size, void* d_ws, size_t ws_size,
                              hipStream_t stream) {
  (void)in_sizes; (void)n_in; (void)out_size; (void)ws_size;
  const float* hidden = (const float*)d_in[0];
  const float* rope   = (const float*)d_in[1];
  const float* attnnw = (const float*)d_in[2];
  const float* ffnnw  = (const float*)d_in[3];
  const float* Wq     = (const float*)d_in[4];
  const float* Wk     = (const float*)d_in[5];
  const float* Wv     = (const float*)d_in[6];
  const float* Wo     = (const float*)d_in[7];
  const float* qnw    = (const float*)d_in[8];
  const float* knw    = (const float*)d_in[9];
  const float* gatew  = (const float*)d_in[10];
  const float* Wg     = (const float*)d_in[11];
  const float* Wu     = (const float*)d_in[12];
  const float* Wd     = (const float*)d_in[13];
  float* out = (float*)d_out;

  // ---- workspace carving ----
  char* wsp = (char*)d_ws;
  auto carve = [&](size_t bytes) -> char* {
    char* p = wsp;
    wsp += (bytes + 255) & ~(size_t)255;
    return p;
  };
  const size_t nDD = (size_t)DM * DM;          // 4.19M
  const size_t nEH = (size_t)EE * DM * MHID;   // 12.58M
  __bf16* WqT = (__bf16*)carve(nDD * 2);       // transposed bf16 mirrors
  __bf16* WkT = (__bf16*)carve(nDD * 2);
  __bf16* WvT = (__bf16*)carve(nDD * 2);
  __bf16* WoT = (__bf16*)carve(nDD * 2);
  __bf16* WgT = (__bf16*)carve(nEH * 2);
  __bf16* WuT = (__bf16*)carve(nEH * 2);
  __bf16* WdT = (__bf16*)carve(nEH * 2);
  __bf16* xb  = (__bf16*)carve((size_t)TT * DM * 2);
  float*  qf  = (float*) carve((size_t)TT * DM * 4);
  float*  kf  = (float*) carve((size_t)TT * DM * 4);
  float*  vf  = (float*) carve((size_t)TT * DM * 4);
  __bf16* qb  = (__bf16*)carve((size_t)TT * DM * 2);
  __bf16* kb  = (__bf16*)carve((size_t)TT * DM * 2);
  __bf16* vT  = (__bf16*)carve((size_t)TT * DM * 2);  // vT[b][h*HD+d][s]
  __bf16* ob  = (__bf16*)carve((size_t)TT * DM * 2);
  __bf16* x2b = (__bf16*)carve((size_t)TT * DM * 2);
  float*  wts = (float*) carve((size_t)TT * EE * 4);
  // Aliases into buffers that are dead by the time these are written:
  float*  hbuf = vf;                                   // residual stream h (f32)
  float*  x2f  = qf;                                   // normed h (f32)
  float*  gbuf = (float*)kf;
  float*  ubuf = (float*)((char*)kf + (size_t)TT * MHID * 4);
  __bf16* hb   = (__bf16*)((char*)kf + (size_t)2 * TT * MHID * 4);

  const int nTD = TT * DM;
  const int nTM = TT * MHID;
  const dim3 tb(32, 8, 1);

  // ---- one bandwidth-bound pass: fp32 weights -> transposed bf16 mirrors ----
  cvt_transpose_kernel<<<dim3(DM / 32, DM / 32, 1), tb, 0, stream>>>(Wq, WqT, DM, DM);
  cvt_transpose_kernel<<<dim3(DM / 32, DM / 32, 1), tb, 0, stream>>>(Wk, WkT, DM, DM);
  cvt_transpose_kernel<<<dim3(DM / 32, DM / 32, 1), tb, 0, stream>>>(Wv, WvT, DM, DM);
  cvt_transpose_kernel<<<dim3(DM / 32, DM / 32, 1), tb, 0, stream>>>(Wo, WoT, DM, DM);
  cvt_transpose_kernel<<<dim3(MHID / 32, DM / 32, EE), tb, 0, stream>>>(Wg, WgT, DM, MHID);
  cvt_transpose_kernel<<<dim3(MHID / 32, DM / 32, EE), tb, 0, stream>>>(Wu, WuT, DM, MHID);
  cvt_transpose_kernel<<<dim3(DM / 32, MHID / 32, EE), tb, 0, stream>>>(Wd, WdT, MHID, DM);

  // ---- attention branch ----
  rmsnorm_kernel<<<TT, 256, 0, stream>>>(hidden, attnnw, xb, (float*)nullptr);
  launch_gemm(xb, WqT, qf, nullptr, nullptr, 0, TT, DM, DM, 0, stream);
  launch_gemm(xb, WkT, kf, nullptr, nullptr, 0, TT, DM, DM, 0, stream);
  launch_gemm(xb, WvT, vf, nullptr, nullptr, 0, TT, DM, DM, 0, stream);
  qknorm_rope_kernel<<<TT * HH, 128, 0, stream>>>(qf, kf, qnw, knw, rope, qb, kb);
  // V: [S, DM] -> transposed bf16 [DM, S] per batch
  cvt_transpose_kernel<<<dim3(DM / 32, SS / 32, BB), tb, 0, stream>>>(vf, vT, SS, DM);
  flash_attn_kernel<<<(BB * HH * (SS / 16) + 3) / 4, 128, 0, stream>>>(qb, kb, vT, ob);
  launch_gemm(ob, WoT, hbuf, hidden, nullptr, 0, TT, DM, DM, 0, stream); // h = hidden + o@Wo

  // ---- MoE branch ----
  rmsnorm_kernel<<<TT, 256, 0, stream>>>(hbuf, ffnnw, x2b, x2f);
  gate_kernel<<<TT, 256, 0, stream>>>(x2f, gatew, wts);
  copy_f32_kernel<<<(nTD + 255) / 256, 256, 0, stream>>>(hbuf, out, nTD); // out = h
  for (int e = 0; e < EE; ++e) {
    launch_gemm(x2b, WgT + (size_t)e * DM * MHID, gbuf, nullptr, nullptr, 0,
                TT, MHID, DM, 0, stream);
    launch_gemm(x2b, WuT + (size_t)e * DM * MHID, ubuf, nullptr, nullptr, 0,
                TT, MHID, DM, 0, stream);
    silu_mul_kernel<<<(nTM + 255) / 256, 256, 0, stream>>>(gbuf, ubuf, hb, nTM);
    // out += weights[:,e] * (hb @ W_down[e])
    launch_gemm(hb, WdT + (size_t)e * MHID * DM, out, nullptr, wts + e, EE,
                TT, DM, MHID, 1, stream);
  }
}
